// DeepseekV2MLAAttention_9053791060142
// MI455X (gfx1250) — matrix operand order repaired and checked by reading, not compiled, not run on hardware
//
#include <hip/hip_runtime.h>

// ---------------- types ----------------
typedef __attribute__((ext_vector_type(16))) __bf16 v16bf;
typedef __attribute__((ext_vector_type(8)))  __bf16 v8bf;
typedef __attribute__((ext_vector_type(8)))  float  v8f;

// ---------------- problem constants ----------------
#define MDIM 8192
#define KDIM 7168
#define NDIM 2112          // 1536 + 512 + 64
#define QLR  1536
#define KVLR 512
#define PEDIM 64
#define NGRP 12            // 1536 / 128

// output offsets (flat concat of the reference's 4-tuple)
#define OFF_QQ   ((size_t)0)
#define OFF_QS   ((size_t)MDIM * QLR)
#define OFF_KV   (OFF_QS + (size_t)MDIM * NGRP)
#define OFF_PE   (OFF_KV + (size_t)MDIM * KVLR)

// ---------------- GEMM tiling ----------------
#define M_TILE 128
#define N_TILE 192
#define K_TILE 32
#define K_PAD  40          // bf16 elems per LDS row (80B stride, 16B aligned)
#define NITER  (KDIM / K_TILE)

// ---------------- helpers ----------------
// 8 floats -> 8 bf16 (native v_cvt path) packed as uint4 for one b128 LDS store
__device__ __forceinline__ uint4 cvt8(float f0, float f1, float f2, float f3,
                                      float f4, float f5, float f6, float f7) {
    v8bf v;
    v[0] = (__bf16)f0; v[1] = (__bf16)f1; v[2] = (__bf16)f2; v[3] = (__bf16)f3;
    v[4] = (__bf16)f4; v[5] = (__bf16)f5; v[6] = (__bf16)f6; v[7] = (__bf16)f7;
    return __builtin_bit_cast(uint4, v);
}
__device__ __forceinline__ v16bf frag_cat(const __bf16* lo, const __bf16* hi) {
    v8bf a = *(const v8bf*)lo;
    v8bf b = *(const v8bf*)hi;
    return __builtin_shufflevector(a, b, 0,1,2,3,4,5,6,7,8,9,10,11,12,13,14,15);
}

// =====================================================================
// Kernel 1: C(8192x2112) = A(8192x7168) @ B(7168x2112), bf16 WMMA, f32 acc
// grid = (11, 64), block = 256 (8 wave32), double-buffered LDS
// =====================================================================
__global__ __launch_bounds__(256) void gemm_bf16(const float* __restrict__ A,
                                                 const float* __restrict__ B,
                                                 float* __restrict__ C) {
    __shared__ __align__(16) __bf16 Asl[2][M_TILE][K_PAD];
    __shared__ __align__(16) __bf16 Bsl[2][N_TILE][K_PAD];   // transposed: [n][k]

    const int t    = threadIdx.x;
    const int lane = t & 31;
    const int wave = t >> 5;
    const int mW   = wave & 3;           // 4 wave rows  -> 32 rows each
    const int nW   = wave >> 2;          // 2 wave cols  -> 96 cols each

    const int colBase = blockIdx.x * N_TILE;          // 0..1920
    const size_t rowBase = (size_t)blockIdx.y * M_TILE;

    // A global mapping: 2 threads per row, 16 floats each
    const int aRow = t >> 1;
    const int aCol = (t & 1) * 16;
    const float* aPtr = A + (rowBase + (size_t)aRow) * KDIM + aCol;

    // B global mapping: threads 0..191 each own one output column (coalesced per k)
    const float* bPtr = B + colBase + (t < N_TILE ? t : 0);

    v8f acc[2][6];
#pragma unroll
    for (int r = 0; r < 2; ++r)
#pragma unroll
        for (int c = 0; c < 6; ++c) acc[r][c] = (v8f)(0.0f);

    const int fm = lane & 15;
    const int kh = lane >> 4;            // 0 or 1

    // ---- prologue: stage K-tile 0 into registers ----
    float4 av[4];
    float  bv[K_TILE];
#pragma unroll
    for (int i = 0; i < 4; ++i)
        av[i] = *(const float4*)(aPtr + i * 4);
    if (t < N_TILE) {
#pragma unroll
        for (int i = 0; i < K_TILE; ++i)
            bv[i] = bPtr[(size_t)i * NDIM];
    }

    for (int it = 0; it < NITER; ++it) {
        const int buf = it & 1;

        // ---- publish staged registers into LDS[buf] ----
        {
            uint4* dst = (uint4*)&Asl[buf][aRow][aCol];
            dst[0] = cvt8(av[0].x, av[0].y, av[0].z, av[0].w,
                          av[1].x, av[1].y, av[1].z, av[1].w);
            dst[1] = cvt8(av[2].x, av[2].y, av[2].z, av[2].w,
                          av[3].x, av[3].y, av[3].z, av[3].w);
        }
        if (t < N_TILE) {
            uint4* dst = (uint4*)&Bsl[buf][t][0];
#pragma unroll
            for (int q = 0; q < 4; ++q)
                dst[q] = cvt8(bv[q*8+0], bv[q*8+1], bv[q*8+2], bv[q*8+3],
                              bv[q*8+4], bv[q*8+5], bv[q*8+6], bv[q*8+7]);
        }

        // ---- issue next tile's global loads (overlap barrier + WMMA) ----
        if (it + 1 < NITER) {
            const int kt = (it + 1) * K_TILE;
#pragma unroll
            for (int i = 0; i < 4; ++i)
                av[i] = *(const float4*)(aPtr + kt + i * 4);
            if (t < N_TILE) {
#pragma unroll
                for (int i = 0; i < K_TILE; ++i)
                    bv[i] = bPtr[(size_t)(kt + i) * NDIM];
            }
        }

        __syncthreads();

        // ---- fragments from LDS[buf] ----
        // A 16x32 bf16 layout: lane<16 -> K 0..7 & 16..23 ; lane>=16 -> +8
        v16bf afrag[2];
#pragma unroll
        for (int r = 0; r < 2; ++r) {
            const __bf16* arow = &Asl[buf][mW * 32 + r * 16 + fm][0];
            afrag[r] = frag_cat(arow + kh * 8, arow + kh * 8 + 16);
        }
        // B 32x16 bf16 layout: lane<16 -> K 0..7 & 16..23 ; lane>=16 -> +8
        v16bf bfrag[6];
#pragma unroll
        for (int c = 0; c < 6; ++c) {
            const __bf16* brow = &Bsl[buf][nW * 96 + c * 16 + fm][0];
            bfrag[c] = frag_cat(brow + kh * 8, brow + kh * 8 + 16);
        }
#pragma unroll
        for (int r = 0; r < 2; ++r)
#pragma unroll
            for (int c = 0; c < 6; ++c)
                acc[r][c] = __builtin_amdgcn_wmma_f32_16x16x32_bf16(
                    false, afrag[r], false, bfrag[c],
                    (short)0, acc[r][c], false, false);
    }

    // ---- write C (f32) ----
#pragma unroll
    for (int r = 0; r < 2; ++r) {
        const size_t rb = rowBase + (size_t)(mW * 32 + r * 16 + kh * 8);
#pragma unroll
        for (int c = 0; c < 6; ++c) {
            const int n = colBase + nW * 96 + c * 16 + fm;
#pragma unroll
            for (int i = 0; i < 8; ++i)
                C[(rb + i) * NDIM + n] = acc[r][c][i];
        }
    }
}

// =====================================================================
// Kernel 2: per-row RMSNorm (q, kv) + group amax quant + k_pe copy
// grid = 8192, block = 256 (8 wave32)
// =====================================================================
__global__ __launch_bounds__(256) void mla_epilogue(const float* __restrict__ C,
                                                    const float* __restrict__ qw,
                                                    const float* __restrict__ kvw,
                                                    float* __restrict__ out) {
    const int row  = blockIdx.x;
    const int t    = threadIdx.x;
    const int lane = t & 31;
    const int wave = t >> 5;
    const float* crow = C + (size_t)row * NDIM;

    __shared__ float wsum[16];
    __shared__ float sgmax[NGRP];
    __shared__ float snorm[2];

    // ---- load q segment: threads 0..191, 8 contiguous elems each ----
    float qv[8];
    float qsum = 0.0f;
    if (t < 192) {
        const float4* p = (const float4*)(crow + t * 8);
        float4 a = p[0], b = p[1];
        qv[0]=a.x; qv[1]=a.y; qv[2]=a.z; qv[3]=a.w;
        qv[4]=b.x; qv[5]=b.y; qv[6]=b.z; qv[7]=b.w;
#pragma unroll
        for (int i = 0; i < 8; ++i) qsum += qv[i] * qv[i];
    }
    // ---- load kv segment: threads 0..127, 4 contiguous elems each ----
    float kvv[4];
    float ksum = 0.0f;
    if (t < 128) {
        float4 a = *(const float4*)(crow + QLR + t * 4);
        kvv[0]=a.x; kvv[1]=a.y; kvv[2]=a.z; kvv[3]=a.w;
#pragma unroll
        for (int i = 0; i < 4; ++i) ksum += kvv[i] * kvv[i];
    }

    // ---- wave-level sum reduce, then cross-wave via LDS ----
#pragma unroll
    for (int m = 16; m >= 1; m >>= 1) {
        qsum += __shfl_xor(qsum, m, 32);
        ksum += __shfl_xor(ksum, m, 32);
    }
    if (lane == 0) { wsum[wave] = qsum; wsum[8 + wave] = ksum; }
    __syncthreads();
    if (t == 0) {
        float s = 0.0f;
#pragma unroll
        for (int w = 0; w < 6; ++w) s += wsum[w];
        snorm[0] = rsqrtf(s / (float)QLR + 1e-6f);
        float k = 0.0f;
#pragma unroll
        for (int w = 0; w < 4; ++w) k += wsum[8 + w];
        snorm[1] = rsqrtf(k / (float)KVLR + 1e-6f);
    }
    __syncthreads();

    // ---- normalize q, per-group (128) amax via 16-lane reduction ----
    float qn[8];
    float am = 0.0f;
    if (t < 192) {
        const float rn = snorm[0];
#pragma unroll
        for (int i = 0; i < 8; ++i) {
            qn[i] = qv[i] * rn * qw[t * 8 + i];
            am = fmaxf(am, fabsf(qn[i]));
        }
#pragma unroll
        for (int m = 1; m <= 8; m <<= 1)
            am = fmaxf(am, __shfl_xor(am, m, 32));
    }
    if (t < 192 && (lane & 15) == 0) sgmax[t >> 4] = am;
    __syncthreads();

    // ---- outputs ----
    if (t < 192) {
        const int g = t >> 4;
        const float sc = fmaxf(sgmax[g] * (1.0f / 448.0f), 1e-12f);
        const float inv = 1.0f / sc;
#pragma unroll
        for (int i = 0; i < 8; ++i)
            out[OFF_QQ + (size_t)row * QLR + t * 8 + i] = qn[i] * inv;
        if ((t & 15) == 0)
            out[OFF_QS + (size_t)row * NGRP + g] = sc;
    }
    if (t < 128) {
        const float rk = snorm[1];
#pragma unroll
        for (int i = 0; i < 4; ++i)
            out[OFF_KV + (size_t)row * KVLR + t * 4 + i] = kvv[i] * rk * kvw[t * 4 + i];
    }
    if (t < PEDIM)
        out[OFF_PE + (size_t)row * PEDIM + t] = crow[QLR + KVLR + t];
}

// =====================================================================
extern "C" void kernel_launch(void* const* d_in, const int* in_sizes, int n_in,
                              void* d_out, int out_size, void* d_ws, size_t ws_size,
                              hipStream_t stream) {
    const float* hs  = (const float*)d_in[0];   // (8192, 7168)
    const float* w   = (const float*)d_in[1];   // (7168, 2112)
    const float* qw  = (const float*)d_in[2];   // (1536,)
    const float* kvw = (const float*)d_in[3];   // (512,)
    float* out = (float*)d_out;
    float* qkv = (float*)d_ws;                  // 8192*2112*4 = ~69.2 MB scratch

    dim3 grid(NDIM / N_TILE, MDIM / M_TILE);    // (11, 64): x-fastest -> A stripe stays in L2
    gemm_bf16<<<grid, 256, 0, stream>>>(hs, w, qkv);
    mla_epilogue<<<MDIM, 256, 0, stream>>>(qkv, qw, kvw, out);
}
